// InterViewDiffusion_45698452030227
// MI455X (gfx1250) — compile-verified, matrix operand order checked by reading, not run on hardware
//
#include <hip/hip_runtime.h>

// ---------------------------------------------------------------------------
// InterViewDiffusion for MI455X (gfx1250, wave32, WMMA + TDM)
//
//  0a. zero stats region of d_ws
//  0b. convert X (f32->bf16) and Wq/Wk/Wv (f32->bf16, concatenated) into ws
//  1.  fused QKV GEMM: Y{q,k,v} = X @ W^T + b via v_wmma_f32_16x16x32_bf16.
//      Tiles staged global->LDS by the Tensor Data Mover
//      (tensor_load_to_lds, double buffered, s_wait_tensorcnt), with D#
//      pad_enable generating the 36-bf16 LDS row stride in hardware.
//      Per-(view,channel) sum/sumsq accumulated via ds_add_f32 + global atomics.
//  2.  BN prep: per-channel affine a,d ; per-view Frobenius^2 analytically
//  3.  correlation: C[i][j][c] = sum_n Yq_i[n,c]*Yk_j[n,c]
//  4.  P matrix: sigmoid, dual softmax, 10-iter doubly-stochastic projection
//  5.  output: out[j,n,d] = av_d*(sum_i c_ij*Yv_i[n,d]) + dv_d*csum_j
//
// ws requirement: ~616 MB (Yq/Yk bf16 + Yv f32 + Xbf + Wbf + stats)
// ---------------------------------------------------------------------------

#define VIEWS 6
#define NROWS 20000
#define DDIM  512
#define BN_EPS 1e-5f

typedef __attribute__((ext_vector_type(16))) __bf16         v16bf;
typedef __attribute__((ext_vector_type(16))) unsigned short v16u;
typedef __attribute__((ext_vector_type(8)))  float          v8f;
typedef __attribute__((ext_vector_type(4)))  unsigned int   v4u;
typedef __attribute__((ext_vector_type(8)))  int            v8i;
typedef __attribute__((ext_vector_type(4)))  int            v4i;

union BF16Frag { v16u u; v16bf b; };

__device__ __forceinline__ unsigned short f2bf(float f) {
  unsigned u = __float_as_uint(f);
  unsigned r = u + 0x7FFFu + ((u >> 16) & 1u);   // round-to-nearest-even
  return (unsigned short)(r >> 16);
}
__device__ __forceinline__ float bf2f(unsigned short b) {
  return __uint_as_float(((unsigned)b) << 16);
}

// --- TDM: load a (tile_d0 x tile_d1 [x tile_d2]) tile of 2-byte elements ----
// pad_enable: after every 16 DWORDs (32 bf16 = one tile row) insert 2 DWORDs
// (4 bf16) of LDS padding -> LDS row stride 36 bf16 (bank-conflict-free).
__device__ __forceinline__ void tdm_load_tile(
    unsigned lds_byte_addr, const unsigned short* gptr,
    unsigned tensor_d0, unsigned tensor_d1, unsigned tensor_d2,
    unsigned tile_d0, unsigned tile_d1, unsigned tile_d2,
    unsigned long long stride0, unsigned long long stride1)
{
  unsigned long long ga = (unsigned long long)(size_t)gptr;
  v4u g0;
  g0[0] = 1u;                                            // count=1 (user D#)
  g0[1] = lds_byte_addr;                                 // lds_addr
  g0[2] = (unsigned)(ga & 0xffffffffu);                  // global_addr lo
  g0[3] = (unsigned)((ga >> 32) & 0x01ffffffu)           // global_addr hi
        | 0x80000000u;                                   // type=2 ("image")
  v8i g1;
  g1[0] = (int)((1u << 16)      // data_size = 2 bytes
              | (1u << 20)      // pad_enable
              | (3u << 22)      // pad_interval: 16 DWORDs
              | (1u << 25));    // pad_amount: 2 DWORDs
  g1[1] = (int)((tensor_d0 & 0xffffu) << 16);
  g1[2] = (int)((tensor_d0 >> 16) | ((tensor_d1 & 0xffffu) << 16));
  g1[3] = (int)((tensor_d1 >> 16) | (tile_d0 << 16));
  g1[4] = (int)(tile_d1 | (tile_d2 << 16));
  g1[5] = (int)(stride0 & 0xffffffffu);
  g1[6] = (int)(((stride0 >> 32) & 0xffffu) | ((unsigned)(stride1 & 0xffffu) << 16));
  g1[7] = (int)((stride1 >> 16) & 0xffffffffu);
  v4i g2;
  g2[0] = (int)tensor_d2; g2[1] = 0; g2[2] = 0; g2[3] = 0;
  v4i g3;
  g3[0] = 0; g3[1] = 0; g3[2] = 0; g3[3] = 0;
  v8i g4;
  g4[0] = 0; g4[1] = 0; g4[2] = 0; g4[3] = 0;
  g4[4] = 0; g4[5] = 0; g4[6] = 0; g4[7] = 0;
  // 6-arg form (amdgpu-toolchain / clang-23): (v4u, v8i, v4i, v4i, v8i, i32)
  __builtin_amdgcn_tensor_load_to_lds(g0, g1, g2, g3, g4, 0);
}

// ------------------------------ phase 0a -----------------------------------
__global__ void zero_kernel(float* __restrict__ p, int n) {
  int i = blockIdx.x * 256 + threadIdx.x;
  if (i < n) p[i] = 0.f;
}

// ------------------------------ phase 0b -----------------------------------
__global__ void __launch_bounds__(256)
cvt_x_kernel(const float* __restrict__ X, unsigned short* __restrict__ Xbf) {
  size_t base = ((size_t)blockIdx.x * 256 + threadIdx.x) * 8;
  float4 a = *(const float4*)(X + base);
  float4 b = *(const float4*)(X + base + 4);
  uint4 o;
  o.x = (unsigned)f2bf(a.x) | ((unsigned)f2bf(a.y) << 16);
  o.y = (unsigned)f2bf(a.z) | ((unsigned)f2bf(a.w) << 16);
  o.z = (unsigned)f2bf(b.x) | ((unsigned)f2bf(b.y) << 16);
  o.w = (unsigned)f2bf(b.z) | ((unsigned)f2bf(b.w) << 16);
  *(uint4*)(Xbf + base) = o;
}

__global__ void __launch_bounds__(256)
cvt_w_kernel(const float* __restrict__ Wq, const float* __restrict__ Wk,
             const float* __restrict__ Wv, unsigned short* __restrict__ Wbf) {
  int m   = blockIdx.x >> 7;                 // 128 blocks per matrix
  size_t base = ((size_t)(blockIdx.x & 127) * 256 + threadIdx.x) * 8;
  const float* src = (m == 0) ? Wq : (m == 1) ? Wk : Wv;
  float4 a = *(const float4*)(src + base);
  float4 b = *(const float4*)(src + base + 4);
  uint4 o;
  o.x = (unsigned)f2bf(a.x) | ((unsigned)f2bf(a.y) << 16);
  o.y = (unsigned)f2bf(a.z) | ((unsigned)f2bf(a.w) << 16);
  o.z = (unsigned)f2bf(b.x) | ((unsigned)f2bf(b.y) << 16);
  o.w = (unsigned)f2bf(b.z) | ((unsigned)f2bf(b.w) << 16);
  *(uint4*)(Wbf + (size_t)m * DDIM * DDIM + base) = o;
}

// ------------------------------ phase 1 ------------------------------------
#define BM  64     // rows per block
#define BNC 128    // cols per block
#define BK  32     // k slice (16x16x32 bf16 WMMA)
#define LDSW 36    // LDS row stride in bf16 (set by TDM pad fields)
#define KT  (DDIM / BK)   // 16 k-iterations

__global__ void __launch_bounds__(256)
qkv_gemm_kernel(const unsigned short* __restrict__ Xbf,
                const unsigned short* __restrict__ Wbf,   // [3][512][512] bf16
                const float* __restrict__ lbq, const float* __restrict__ lbk,
                const float* __restrict__ lbv,
                unsigned short* __restrict__ Yq, unsigned short* __restrict__ Yk,
                float* __restrict__ Yv,
                float* __restrict__ gSum, float* __restrict__ gSq)
{
  __shared__ __align__(16) unsigned short sX[2][BM][LDSW];        //  2 x 4.5 KB
  __shared__ __align__(16) unsigned short sW[2][3][BNC][LDSW];    //  2 x 27 KB
  __shared__ float sSum[3][BNC];
  __shared__ float sSq[3][BNC];

  const int tid     = threadIdx.x;
  const int rowBase = blockIdx.x * BM;
  const int colBase = blockIdx.y * BNC;
  const int view    = blockIdx.z;

  for (int i = tid; i < 3 * BNC; i += 256) {
    (&sSum[0][0])[i] = 0.f;
    (&sSq[0][0])[i]  = 0.f;
  }

  const int lane = tid & 31;
  const int wid  = tid >> 5;     // 8 waves
  const int wr   = wid >> 2;     // 0..1 : 32-row strip
  const int wc   = wid & 3;      // 0..3 : 32-col strip
  const int fr   = lane & 15;
  const int kh   = (lane >> 4) & 1;

  v8f acc[3][2][2];
  for (int m = 0; m < 3; ++m)
    for (int i = 0; i < 2; ++i)
      for (int j = 0; j < 2; ++j)
        for (int v = 0; v < 8; ++v) acc[m][i][j][v] = 0.0f;

  const unsigned short* xTileBase = Xbf + ((size_t)view * NROWS + rowBase) * DDIM;
  const unsigned short* wTileBase = Wbf + (size_t)colBase * DDIM;
  const unsigned rowsLeft = (unsigned)(NROWS - rowBase);   // TDM zero-fills OOB

  // issue one X-tile TDM (2D) + one 3-matrix W-tile TDM (3D) for slice kt
  auto issue_kt = [&](int kt, int buf) {
    tdm_load_tile((unsigned)(size_t)&sX[buf][0][0], xTileBase + kt * BK,
                  DDIM, rowsLeft, 0, BK, BM, 0,
                  (unsigned long long)DDIM, 0ull);
    tdm_load_tile((unsigned)(size_t)&sW[buf][0][0][0], wTileBase + kt * BK,
                  DDIM, DDIM, 3, BK, BNC, 3,
                  (unsigned long long)DDIM,
                  (unsigned long long)DDIM * DDIM);
  };

  if (tid < 32) {                 // wave 0 drives the TDM
    issue_kt(0, 0);
    issue_kt(1, 1);
    __builtin_amdgcn_s_wait_tensorcnt((short)2);   // buf0 ready
  }
  __syncthreads();

  for (int kt = 0; kt < KT; ++kt) {
    const int buf = kt & 1;
    // ---- fragments & WMMA from LDS buffer `buf` ----
    BF16Frag aF[2];
    for (int i = 0; i < 2; ++i) {
      int r = wr * 32 + i * 16 + fr;
      for (int v = 0; v < 8; ++v) {
        int kk = ((v >> 2) * 16) + kh * 8 + ((v & 3) * 2);
        aF[i].u[2 * v]     = sX[buf][r][kk];
        aF[i].u[2 * v + 1] = sX[buf][r][kk + 1];
      }
    }
    for (int m = 0; m < 3; ++m) {
      for (int j = 0; j < 2; ++j) {
        BF16Frag bF;
        int c = wc * 32 + j * 16 + fr;
        for (int v = 0; v < 8; ++v) {
          int kk = ((v >> 2) * 16) + kh * 8 + ((v & 3) * 2);
          bF.u[2 * v]     = sW[buf][m][c][kk];
          bF.u[2 * v + 1] = sW[buf][m][c][kk + 1];
        }
        for (int i = 0; i < 2; ++i) {
          acc[m][i][j] = __builtin_amdgcn_wmma_f32_16x16x32_bf16(
              false, aF[i].b, false, bF.b, (short)0, acc[m][i][j], false, false);
        }
      }
    }
    if (kt == KT - 1) break;
    __syncthreads();              // everyone done reading buf `kt&1`
    if (tid < 32) {
      if (kt + 2 < KT) {
        issue_kt(kt + 2, buf);    // refill the buffer just drained
        __builtin_amdgcn_s_wait_tensorcnt((short)2);  // kt+1 loads done
      } else {
        __builtin_amdgcn_s_wait_tensorcnt((short)0);  // last loads done
      }
    }
    __syncthreads();              // buf (kt+1)&1 ready for all waves
  }
  __syncthreads();

  // ---- epilogue: bias add, store Y, LDS stat accumulation ----
  const float* lb[3] = {lbq, lbk, lbv};
  for (int m = 0; m < 3; ++m) {
    for (int j = 0; j < 2; ++j) {
      int lcol = wc * 32 + j * 16 + fr;
      int gcol = colBase + lcol;
      float bias = lb[m][gcol];
      for (int i = 0; i < 2; ++i) {
        float ls = 0.f, lq = 0.f;
        for (int v = 0; v < 8; ++v) {
          int grow = rowBase + wr * 32 + i * 16 + kh * 8 + v;
          if (grow < NROWS) {
            float y = acc[m][i][j][v] + bias;
            size_t idx = ((size_t)view * NROWS + grow) * DDIM + gcol;
            if      (m == 0) Yq[idx] = f2bf(y);
            else if (m == 1) Yk[idx] = f2bf(y);
            else             Yv[idx] = y;
            ls += y;
            lq += y * y;
          }
        }
        atomicAdd(&sSum[m][lcol], ls);   // ds_add_f32
        atomicAdd(&sSq[m][lcol],  lq);
      }
    }
  }
  __syncthreads();
  for (int t = tid; t < 3 * BNC; t += 256) {
    int m = t >> 7, lcol = t & 127;
    int gcol = colBase + lcol;
    atomicAdd(&gSum[(m * VIEWS + view) * DDIM + gcol], sSum[m][lcol]);
    atomicAdd(&gSq [(m * VIEWS + view) * DDIM + gcol], sSq[m][lcol]);
  }
}

// ------------------------------ phase 2 ------------------------------------
__global__ void bn_prep_kernel(const float* __restrict__ gSum, const float* __restrict__ gSq,
                               const float* __restrict__ gq, const float* __restrict__ betaq,
                               const float* __restrict__ gk, const float* __restrict__ betak,
                               const float* __restrict__ gv, const float* __restrict__ betav,
                               float* __restrict__ aC, float* __restrict__ dC,
                               float* __restrict__ F2)
{
  int c = threadIdx.x;  // 512 threads
  const float* gamma[3] = {gq, gk, gv};
  const float* beta[3]  = {betaq, betak, betav};
  const float invM = 1.0f / (float)(VIEWS * NROWS);
  for (int m = 0; m < 3; ++m) {
    float s = 0.f, q = 0.f;
    for (int i = 0; i < VIEWS; ++i) {
      s += gSum[(m * VIEWS + i) * DDIM + c];
      q += gSq [(m * VIEWS + i) * DDIM + c];
    }
    float mean = s * invM;
    float var  = q * invM - mean * mean;      // biased variance
    float a = gamma[m][c] * rsqrtf(var + BN_EPS);
    float d = beta[m][c] - a * mean;
    aC[m * DDIM + c] = a;
    dC[m * DDIM + c] = d;
    for (int i = 0; i < VIEWS; ++i) {
      float Si = gSum[(m * VIEWS + i) * DDIM + c];
      float Qi = gSq [(m * VIEWS + i) * DDIM + c];
      atomicAdd(&F2[m * VIEWS + i],
                a * a * Qi + 2.f * a * d * Si + (float)NROWS * d * d);
    }
  }
}

// ------------------------------ phase 3 ------------------------------------
#define CORR_ROWS 50
__global__ void __launch_bounds__(256)
corr_kernel(const unsigned short* __restrict__ Yq,
            const unsigned short* __restrict__ Yk,
            float* __restrict__ C)
{
  int tid = threadIdx.x;
  int c0  = tid * 2;                 // channel pair per thread (one b32 load)
  int n0  = blockIdx.x * CORR_ROWS;
  float acc[36][2];
  for (int p = 0; p < 36; ++p) { acc[p][0] = 0.f; acc[p][1] = 0.f; }
  for (int r = 0; r < CORR_ROWS; ++r) {
    int n = n0 + r;
    float qv[VIEWS][2], kv[VIEWS][2];
    for (int i = 0; i < VIEWS; ++i) {
      unsigned u = *(const unsigned*)(Yq + ((size_t)i * NROWS + n) * DDIM + c0);
      qv[i][0] = bf2f((unsigned short)u); qv[i][1] = bf2f((unsigned short)(u >> 16));
      unsigned w = *(const unsigned*)(Yk + ((size_t)i * NROWS + n) * DDIM + c0);
      kv[i][0] = bf2f((unsigned short)w); kv[i][1] = bf2f((unsigned short)(w >> 16));
    }
    for (int i = 0; i < VIEWS; ++i)
      for (int j = 0; j < VIEWS; ++j) {
        acc[i * 6 + j][0] += qv[i][0] * kv[j][0];
        acc[i * 6 + j][1] += qv[i][1] * kv[j][1];
      }
  }
  for (int p = 0; p < 36; ++p) {
    atomicAdd(&C[p * DDIM + c0],     acc[p][0]);
    atomicAdd(&C[p * DDIM + c0 + 1], acc[p][1]);
  }
}

// ------------------------------ phase 4 ------------------------------------
__global__ void pmat_kernel(const float* __restrict__ C,
                            const float* __restrict__ gSum,
                            const float* __restrict__ aC, const float* __restrict__ dC,
                            const float* __restrict__ F2,
                            float* __restrict__ coef)   // [0..35]=c_ij, [36..41]=csum_j
{
  __shared__ float red[256];
  __shared__ float sL[36];
  int tid = threadIdx.x;
  for (int p = 0; p < 36; ++p) {
    int i = p / 6, j = p % 6;
    float part = 0.f;
    for (int c = tid; c < DDIM; c += 256) {
      float aq = aC[c],            dq = dC[c];
      float ak = aC[DDIM + c],     dk = dC[DDIM + c];
      float Sq = gSum[(0 * VIEWS + i) * DDIM + c];
      float Sk = gSum[(1 * VIEWS + j) * DDIM + c];
      part += aq * ak * C[p * DDIM + c] + aq * dk * Sq + dq * ak * Sk
            + (float)NROWS * dq * dk;
    }
    red[tid] = part;
    __syncthreads();
    for (int s = 128; s > 0; s >>= 1) {
      if (tid < s) red[tid] += red[tid + s];
      __syncthreads();
    }
    if (tid == 0) sL[p] = red[0];
    __syncthreads();
  }
  if (tid == 0) {
    float P[6][6], S0[6][6], S1[6][6], Fq[6], Fk[6], Fv[6];
    for (int i = 0; i < 6; ++i) {
      Fq[i] = sqrtf(F2[i])      + 1e-8f;
      Fk[i] = sqrtf(F2[6 + i])  + 1e-8f;
      Fv[i] = sqrtf(F2[12 + i]) + 1e-8f;
    }
    for (int i = 0; i < 6; ++i)
      for (int j = 0; j < 6; ++j) {
        float x = sL[i * 6 + j] / (Fq[i] * Fk[j]);
        P[i][j] = 1.f / (1.f + expf(-x));
      }
    for (int j = 0; j < 6; ++j) {            // softmax over axis 0
      float mx = -1e30f;
      for (int i = 0; i < 6; ++i) mx = fmaxf(mx, P[i][j]);
      float s = 0.f;
      for (int i = 0; i < 6; ++i) { S0[i][j] = expf(P[i][j] - mx); s += S0[i][j]; }
      for (int i = 0; i < 6; ++i) S0[i][j] /= s;
    }
    for (int i = 0; i < 6; ++i) {            // softmax over axis 1
      float mx = -1e30f;
      for (int j = 0; j < 6; ++j) mx = fmaxf(mx, P[i][j]);
      float s = 0.f;
      for (int j = 0; j < 6; ++j) { S1[i][j] = expf(P[i][j] - mx); s += S1[i][j]; }
      for (int j = 0; j < 6; ++j) S1[i][j] /= s;
    }
    for (int i = 0; i < 6; ++i)
      for (int j = 0; j < 6; ++j) P[i][j] = 0.5f * (S0[i][j] + S1[i][j]);
    for (int it = 0; it < 10; ++it) {
      for (int i = 0; i < 6; ++i)
        for (int j = 0; j < 6; ++j) P[i][j] = fmaxf(P[i][j], 0.f);
      for (int i = 0; i < 6; ++i) {
        float rs = 0.f;
        for (int j = 0; j < 6; ++j) rs += P[i][j];
        float adj = (rs - 1.f) * (1.f / 6.f);
        for (int j = 0; j < 6; ++j) P[i][j] -= adj;
      }
      for (int j = 0; j < 6; ++j) {
        float cs = 0.f;
        for (int i = 0; i < 6; ++i) cs += P[i][j];
        float adj = (cs - 1.f) * (1.f / 6.f);
        for (int i = 0; i < 6; ++i) P[i][j] -= adj;
      }
    }
    for (int j = 0; j < 6; ++j) {
      float cs = 0.f;
      for (int i = 0; i < 6; ++i) {
        float cij = P[i][j] / Fv[i];
        coef[i * 6 + j] = cij;
        cs += cij;
      }
      coef[36 + j] = cs;
    }
  }
}

// ------------------------------ phase 5 ------------------------------------
__global__ void __launch_bounds__(256)
out_kernel(const float* __restrict__ Yv,
           const float* __restrict__ aC, const float* __restrict__ dC,
           const float* __restrict__ coef,
           float* __restrict__ out)
{
  size_t idx = (size_t)blockIdx.x * 256 + threadIdx.x;  // over NROWS * (DDIM/4)
  int n = (int)(idx >> 7);
  int d = ((int)idx & 127) * 4;
  float4 av = *(const float4*)(aC + 2 * DDIM + d);
  float4 dv = *(const float4*)(dC + 2 * DDIM + d);
  float4 yv[VIEWS];
  for (int i = 0; i < VIEWS; ++i)
    yv[i] = *(const float4*)(Yv + ((size_t)i * NROWS + n) * DDIM + d);
  for (int j = 0; j < VIEWS; ++j) {
    float cs = coef[36 + j];
    float4 s = {0.f, 0.f, 0.f, 0.f};
    for (int i = 0; i < VIEWS; ++i) {
      float cc = coef[i * 6 + j];
      s.x += cc * yv[i].x; s.y += cc * yv[i].y;
      s.z += cc * yv[i].z; s.w += cc * yv[i].w;
    }
    float4 o;
    o.x = av.x * s.x + dv.x * cs;  o.y = av.y * s.y + dv.y * cs;
    o.z = av.z * s.z + dv.z * cs;  o.w = av.w * s.w + dv.w * cs;
    *(float4*)(out + ((size_t)j * NROWS + n) * DDIM + d) = o;
  }
}

// ------------------------------ launch -------------------------------------
extern "C" void kernel_launch(void* const* d_in, const int* in_sizes, int n_in,
                              void* d_out, int out_size, void* d_ws, size_t ws_size,
                              hipStream_t stream)
{
  const float* X    = (const float*)d_in[0];
  const float* Wq   = (const float*)d_in[1];
  const float* lbq  = (const float*)d_in[2];
  const float* Wk   = (const float*)d_in[3];
  const float* lbk  = (const float*)d_in[4];
  const float* Wv   = (const float*)d_in[5];
  const float* lbv  = (const float*)d_in[6];
  const float* gq   = (const float*)d_in[7];
  const float* betq = (const float*)d_in[8];
  const float* gk   = (const float*)d_in[9];
  const float* betk = (const float*)d_in[10];
  const float* gv   = (const float*)d_in[11];
  const float* betv = (const float*)d_in[12];
  float* out = (float*)d_out;

  const size_t NY = (size_t)VIEWS * NROWS * DDIM;  // 61,440,000 elems
  char* ws = (char*)d_ws;
  unsigned short* Yq  = (unsigned short*)(ws);                 // bf16
  unsigned short* Yk  = (unsigned short*)(ws + 2 * NY);        // bf16
  float*          Yv  = (float*)(ws + 4 * NY);                 // f32
  unsigned short* Xbf = (unsigned short*)(ws + 8 * NY);        // bf16
  unsigned short* Wbf = (unsigned short*)(ws + 10 * NY);       // bf16 [3][512][512]
  char* stats = ws + 10 * NY + (size_t)3 * DDIM * DDIM * 2;
  float* gSum = (float*)(stats);                               // 3*6*512
  float* gSq  = (float*)(stats + 36864);                       // 3*6*512
  float* aC   = (float*)(stats + 73728);                       // 3*512
  float* dC   = (float*)(stats + 79872);                       // 3*512
  float* F2   = (float*)(stats + 86016);                       // 18 (pad 256)
  float* Cbuf = (float*)(stats + 86272);                       // 36*512
  float* coef = (float*)(stats + 160000);                      // 42

  // phase 0a: zero stats region (gSum..coef)
  {
    int nz = 160064 / 4 + 64;
    zero_kernel<<<(nz + 255) / 256, 256, 0, stream>>>(gSum, nz);
  }
  // phase 0b: bf16 conversions (X: 61.44M elems, 8/thread; W: 3*262144 elems)
  cvt_x_kernel<<<(unsigned)(NY / (256 * 8)), 256, 0, stream>>>(X, Xbf);
  cvt_w_kernel<<<384, 256, 0, stream>>>(Wq, Wk, Wv, Wbf);
  // phase 1: fused QKV WMMA GEMM (TDM-staged) + stats
  {
    dim3 grid((NROWS + BM - 1) / BM, DDIM / BNC, VIEWS);  // 313 x 4 x 6
    qkv_gemm_kernel<<<grid, 256, 0, stream>>>(Xbf, Wbf, lbq, lbk, lbv,
                                              Yq, Yk, Yv, gSum, gSq);
  }
  // phase 2: BN coefficients + Frobenius^2
  bn_prep_kernel<<<1, DDIM, 0, stream>>>(gSum, gSq, gq, betq, gk, betk, gv, betv,
                                         aC, dC, F2);
  // phase 3: per-channel Q/K cross-view correlation
  corr_kernel<<<NROWS / CORR_ROWS, 256, 0, stream>>>(Yq, Yk, Cbuf);
  // phase 4: P matrix + doubly-stochastic projection + mixing coefficients
  pmat_kernel<<<1, 256, 0, stream>>>(Cbuf, gSum, aC, dC, F2, coef);
  // phase 5: output mixing
  out_kernel<<<(unsigned)(NROWS * (DDIM / 4) / 256), 256, 0, stream>>>(Yv, aC, dC,
                                                                       coef, out);
}
